// BG_ALRT_5574867550257
// MI455X (gfx1250) — compile-verified, matrix operand order checked by reading, not run on hardware
//
#include <hip/hip_runtime.h>
#include <hip/hip_bf16.h>

// ---------------------------------------------------------------------------
// Model dims (fixed by the reference)
// ---------------------------------------------------------------------------
#define TT      512
#define EMB     1024
#define GDIM    128
#define NGRP    8
#define NLAY    12
#define NNODES  96
#define O3      384
#define O4      512
#define NVOCAB  50257
#define NSTEPS  8
#define VTILES  3142   // ceil(50257/16)
#define VQUADS  786    // ceil(3142/4)

typedef __bf16 v16bf __attribute__((ext_vector_type(16)));
typedef __bf16 v8bf  __attribute__((ext_vector_type(8)));
typedef float  v8f   __attribute__((ext_vector_type(8)));
typedef unsigned u32x4 __attribute__((ext_vector_type(4)));
typedef int      i32x8 __attribute__((ext_vector_type(8)));
typedef int      i32x4 __attribute__((ext_vector_type(4)));

#define DEV static __device__ __forceinline__

// hardware RNE conversion (v_cvt_pk_bf16_f32 when pairable)
DEV __bf16 f2b(float x) { return (__bf16)x; }

DEV v8f wmma_bf16(v16bf a, v16bf b, v8f c) {
  return __builtin_amdgcn_wmma_f32_16x16x32_bf16(false, a, false, b,
                                                 (short)0, c, false, false);
}

// 16x32 WMMA operand tile (A layout; identical layout serves B with N as rows)
// elem i of lane L -> (m0 + (L&15),  k0 + 8*(L>>4) + (i&7) + 16*(i>>3))
DEV v16bf ld_bf16_tile(const __bf16* __restrict__ base, int stride,
                       int m0, int k0, int lane) {
  int m = m0 + (lane & 15);
  int k = k0 + ((lane >> 4) << 3);
  v8bf lo = *(const v8bf*)(base + (size_t)m * stride + k);
  v8bf hi = *(const v8bf*)(base + (size_t)m * stride + k + 16);
  v16bf r;
#pragma unroll
  for (int i = 0; i < 8; ++i) { r[i] = lo[i]; r[8 + i] = hi[i]; }
  return r;
}

DEV float red16_sum(float v) {
#pragma unroll
  for (int m = 1; m < 16; m <<= 1) v += __shfl_xor(v, m, 32);
  return v;
}
DEV float red32_sum(float v) {
#pragma unroll
  for (int m = 1; m < 32; m <<= 1) v += __shfl_xor(v, m, 32);
  return v;
}

// ---------------------------------------------------------------------------
// Setup kernels
// ---------------------------------------------------------------------------

// out[row] = sum_k in[row*K+k]   (rowsum of attn_proj / mlp_proj)
__global__ void rowsum_kernel(const float* __restrict__ in,
                              float* __restrict__ out, int K) {
  int row = blockIdx.x, tid = threadIdx.x;
  float s = 0.f;
  for (int k = tid; k < K; k += 128) s += in[(size_t)row * K + k];
  __shared__ float sr[4];
  s = red32_sum(s);
  if ((tid & 31) == 0) sr[tid >> 5] = s;
  __syncthreads();
  if (tid == 0) out[row] = sr[0] + sr[1] + sr[2] + sr[3];
}

// depths = iterate 12x: relu(D) @ (depths+1);  wm[t][n] precomputed for 8 steps
__global__ void depth_kernel(const float* __restrict__ D,
                             float* __restrict__ depths,
                             float* __restrict__ wm) {
  __shared__ float cur[NNODES], nxt[NNODES];
  int tid = threadIdx.x;
  if (tid < NNODES) cur[tid] = 0.f;
  __syncthreads();
  for (int it = 0; it < NLAY; ++it) {
    if (tid < NNODES) {
      float s = 0.f;
      for (int m = 0; m < NNODES; ++m) {
        float d = D[tid * NNODES + m];
        d = d > 0.f ? d : 0.f;
        s += d * (cur[m] + 1.f);
      }
      nxt[tid] = s;
    }
    __syncthreads();
    if (tid < NNODES) cur[tid] = nxt[tid];
    __syncthreads();
  }
  if (tid < NNODES) depths[tid] = cur[tid];
  for (int i = tid; i < NSTEPS * NNODES; i += blockDim.x) {
    int t = i / NNODES, n = i % NNODES;
    float td = (float)t * 1.5f;  // N_LAYER / N_STEPS
    float w = __expf(-fabsf(cur[n] - td));
    wm[i] = w > 0.15f ? w : 0.f;
  }
}

// x[t,:] = rmsnorm(wte[idx[t],:]) ; p_cont[t] = 1
__global__ void embed_kernel(const int* __restrict__ idx,
                             const float* __restrict__ wte,
                             float* __restrict__ x, float* __restrict__ pcont) {
  int t = blockIdx.x, tid = threadIdx.x;  // 256 threads
  const float* row = wte + (size_t)idx[t] * EMB;
  float v[4], ss = 0.f;
#pragma unroll
  for (int j = 0; j < 4; ++j) { v[j] = row[tid + 256 * j]; ss += v[j] * v[j]; }
  __shared__ float sr[8];
  ss = red32_sum(ss);
  if ((tid & 31) == 0) sr[tid >> 5] = ss;
  __syncthreads();
  float tot = 0.f;
#pragma unroll
  for (int i = 0; i < 8; ++i) tot += sr[i];
  float r = rsqrtf(tot / (float)EMB + 1e-6f);
#pragma unroll
  for (int j = 0; j < 4; ++j) x[(size_t)t * EMB + tid + 256 * j] = v[j] * r;
  if (tid == 0) pcont[t] = 1.f;
}

// generic f32 -> bf16 conversion (x and the per-launch weight preconversions)
__global__ void f2b_kernel(const float* __restrict__ in, __bf16* __restrict__ o) {
  int i = blockIdx.x * 256 + threadIdx.x;
  o[i] = f2b(in[i]);
}

// ---------------------------------------------------------------------------
// Per-step kernels
// ---------------------------------------------------------------------------

// qkv[n,t,o] = xb[t, (n%8)*128 + g] @ qkv_wb[n,o,g]^T  (one 16x16 tile / wave)
__global__ void qkv_kernel(const __bf16* __restrict__ xb,
                           const __bf16* __restrict__ qkv_wb,
                           float* __restrict__ qkv) {
  int wid = blockIdx.x * 4 + (threadIdx.x >> 5);
  int lane = threadIdx.x & 31;
  int node = wid / 768;            // 32 t-tiles * 24 o-tiles
  int rem  = wid % 768;
  int tt = rem / 24, ot = rem % 24;
  const __bf16* A = xb + (node & 7) * GDIM;
  const __bf16* B = qkv_wb + (size_t)node * O3 * GDIM;
  v8f acc = {0, 0, 0, 0, 0, 0, 0, 0};
#pragma unroll
  for (int k0 = 0; k0 < GDIM; k0 += 32) {
    v16bf a = ld_bf16_tile(A, EMB, tt * 16, k0, lane);
    v16bf b = ld_bf16_tile(B, GDIM, ot * 16, k0, lane);
    acc = wmma_bf16(a, b, acc);
  }
  float* C = qkv + (size_t)node * TT * O3;
  int col = ot * 16 + (lane & 15);
  int r0  = tt * 16 + ((lane >> 4) << 3);
#pragma unroll
  for (int r = 0; r < 8; ++r) C[(size_t)(r0 + r) * O3 + col] = acc[r];
}

// rotary + rmsnorm on q,k ; write qb/kb bf16 [n][t][d] and v transposed
// vtb bf16 [n][d][t].  One wave per (node,t) row.
__global__ void rot_kernel(const float* __restrict__ qkv,
                           __bf16* __restrict__ qb, __bf16* __restrict__ kb,
                           __bf16* __restrict__ vtb) {
  int w = blockIdx.x * 4 + (threadIdx.x >> 5);
  int lane = threadIdx.x & 31;
  int node = w >> 9, t = w & 511;
  const float* base = qkv + (size_t)(node * TT + t) * O3;
  float q[4], k[4], v[4];
#pragma unroll
  for (int i = 0; i < 4; ++i) {
    int d = lane + 32 * i;
    q[i] = base[d]; k[i] = base[GDIM + d]; v[i] = base[2 * GDIM + d];
  }
  float a0 = (float)t * __powf(10000.f, -(float)lane / 64.f);
  float a1 = (float)t * __powf(10000.f, -(float)(lane + 32) / 64.f);
  float c0 = __cosf(a0), s0 = __sinf(a0), c1 = __cosf(a1), s1 = __sinf(a1);
  float rq[4], rk[4];
  rq[0] =  q[0] * c0 + q[2] * s0;  rq[2] = -q[0] * s0 + q[2] * c0;
  rq[1] =  q[1] * c1 + q[3] * s1;  rq[3] = -q[1] * s1 + q[3] * c1;
  rk[0] =  k[0] * c0 + k[2] * s0;  rk[2] = -k[0] * s0 + k[2] * c0;
  rk[1] =  k[1] * c1 + k[3] * s1;  rk[3] = -k[1] * s1 + k[3] * c1;
  float sq = 0.f, sk = 0.f;
#pragma unroll
  for (int i = 0; i < 4; ++i) { sq += rq[i] * rq[i]; sk += rk[i] * rk[i]; }
  sq = red32_sum(sq); sk = red32_sum(sk);
  float nq = rsqrtf(sq / (float)GDIM + 1e-6f);
  float nk = rsqrtf(sk / (float)GDIM + 1e-6f);
  __bf16* qr = qb + (size_t)(node * TT + t) * GDIM;
  __bf16* kr = kb + (size_t)(node * TT + t) * GDIM;
#pragma unroll
  for (int i = 0; i < 4; ++i) {
    int d = lane + 32 * i;
    qr[d] = f2b(rq[i] * nq);
    kr[d] = f2b(rk[i] * nk);
    vtb[((size_t)node * GDIM + d) * TT + t] = f2b(v[i]);
  }
}

// Flash attention, one wave per (node, 16-query tile).
// S^T = K.Q^T so the D-layout of the score WMMA is directly the B operand
// of out^T = V^T.P (lane-local repack, no LDS transpose).
__global__ void attn_kernel(const __bf16* __restrict__ qb,
                            const __bf16* __restrict__ kb,
                            const __bf16* __restrict__ vtb,
                            float* __restrict__ att) {
  int w = blockIdx.x * 4 + (threadIdx.x >> 5);   // 96*32 waves
  int lane = threadIdx.x & 31;
  int node = w >> 5, qt = w & 31;
  int qbase = qt * 16;
  const __bf16* Q  = qb  + (size_t)node * TT * GDIM;
  const __bf16* K  = kb  + (size_t)node * TT * GDIM;
  const __bf16* VT = vtb + (size_t)node * GDIM * TT;

  v16bf Bq[4];
#pragma unroll
  for (int db = 0; db < 4; ++db) Bq[db] = ld_bf16_tile(Q, GDIM, qbase, db * 32, lane);

  v8f out[8];
#pragma unroll
  for (int dt = 0; dt < 8; ++dt) out[dt] = (v8f){0, 0, 0, 0, 0, 0, 0, 0};

  float m = -1e30f, l = 0.f;
  int qcol = qbase + (lane & 15);
  int hi8  = (lane >> 4) << 3;
  const float scale = 0.0883883476f;  // 1/sqrt(128)

  for (int kb0 = 0; kb0 <= qbase + 15; kb0 += 32) {
    v8f s0 = (v8f){0,0,0,0,0,0,0,0}, s1 = (v8f){0,0,0,0,0,0,0,0};
#pragma unroll
    for (int db = 0; db < 4; ++db) {
      v16bf ak0 = ld_bf16_tile(K, GDIM, kb0,      db * 32, lane);
      v16bf ak1 = ld_bf16_tile(K, GDIM, kb0 + 16, db * 32, lane);
      s0 = wmma_bf16(ak0, Bq[db], s0);
      s1 = wmma_bf16(ak1, Bq[db], s1);
    }
    float lm = -1e30f, p0[8], p1[8];
#pragma unroll
    for (int r = 0; r < 8; ++r) {
      int key0 = kb0 + r + hi8;
      float v0 = (key0      <= qcol) ? s0[r] * scale : -1e30f;
      float v1 = (key0 + 16 <= qcol) ? s1[r] * scale : -1e30f;
      p0[r] = v0; p1[r] = v1;
      lm = fmaxf(lm, fmaxf(v0, v1));
    }
    lm = fmaxf(lm, __shfl_xor(lm, 16, 32));   // merge two key half-ranges
    float mn = fmaxf(m, lm);
    float corr = __expf(m - mn);
    m = mn;
    float rs = 0.f;
    v16bf pb;
#pragma unroll
    for (int r = 0; r < 8; ++r) {
      float e0 = __expf(p0[r] - mn), e1 = __expf(p1[r] - mn);
      rs += e0 + e1;
      pb[r] = f2b(e0); pb[8 + r] = f2b(e1);
    }
    rs += __shfl_xor(rs, 16, 32);
    l = l * corr + rs;
#pragma unroll
    for (int dt = 0; dt < 8; ++dt) {
#pragma unroll
      for (int r = 0; r < 8; ++r) out[dt][r] *= corr;
      v16bf av = ld_bf16_tile(VT, TT, dt * 16, kb0, lane);
      out[dt] = wmma_bf16(av, pb, out[dt]);
    }
  }
  float inv = 1.f / l;
  float* arow = att + (size_t)node * TT * GDIM;
#pragma unroll
  for (int dt = 0; dt < 8; ++dt)
#pragma unroll
    for (int r = 0; r < 8; ++r)
      arow[(size_t)qcol * GDIM + dt * 16 + r + hi8] = out[dt][r] * inv;
}

// nb[n,t,g] = rmsnorm( x[t,(n%8)*128+g] + att[n,t,g]*aps[n,g] )  (bf16)
__global__ void midnorm_kernel(const float* __restrict__ x,
                               const float* __restrict__ att,
                               const float* __restrict__ aps,
                               __bf16* __restrict__ nb) {
  int w = blockIdx.x * 4 + (threadIdx.x >> 5);
  int lane = threadIdx.x & 31;
  int node = w >> 9, t = w & 511;
  const float* xr = x + (size_t)t * EMB + (node & 7) * GDIM;
  const float* ar = att + (size_t)(node * TT + t) * GDIM;
  const float* ap = aps + node * GDIM;
  float v[4], ss = 0.f;
#pragma unroll
  for (int i = 0; i < 4; ++i) {
    int d = lane + 32 * i;
    v[i] = xr[d] + ar[d] * ap[d];
    ss += v[i] * v[i];
  }
  ss = red32_sum(ss);
  float r = rsqrtf(ss / (float)GDIM + 1e-6f);
  __bf16* o = nb + (size_t)(node * TT + t) * GDIM;
#pragma unroll
  for (int i = 0; i < 4; ++i) o[lane + 32 * i] = f2b(v[i] * r);
}

// fcsum[n,t] = sum_o relu(nb[n,t,:] @ mlp_fcb[n,o,:]^T)^2 ; wave per (node,ttile)
__global__ void fc_kernel(const __bf16* __restrict__ nb,
                          const __bf16* __restrict__ mlp_fcb,
                          float* __restrict__ fcsum) {
  int w = blockIdx.x * 4 + (threadIdx.x >> 5);
  int lane = threadIdx.x & 31;
  int node = w >> 5, tt = w & 31;
  const __bf16* A = nb + (size_t)node * TT * GDIM;
  const __bf16* B = mlp_fcb + (size_t)node * O4 * GDIM;
  v16bf a[4];
#pragma unroll
  for (int db = 0; db < 4; ++db) a[db] = ld_bf16_tile(A, GDIM, tt * 16, db * 32, lane);
  float ps[8] = {0, 0, 0, 0, 0, 0, 0, 0};
  for (int ot = 0; ot < 32; ++ot) {
    v8f acc = (v8f){0, 0, 0, 0, 0, 0, 0, 0};
#pragma unroll
    for (int db = 0; db < 4; ++db) {
      v16bf b = ld_bf16_tile(B, GDIM, ot * 16, db * 32, lane);
      acc = wmma_bf16(a[db], b, acc);
    }
#pragma unroll
    for (int r = 0; r < 8; ++r) {
      float t0 = fmaxf(acc[r], 0.f);
      ps[r] += t0 * t0;
    }
  }
#pragma unroll
  for (int r = 0; r < 8; ++r) {
    float s = red16_sum(ps[r]);
    if ((lane & 15) == 0)
      fcsum[node * TT + tt * 16 + r + ((lane >> 4) << 3)] = s;
  }
}

// x += sum_l (att*aps + fcsum*mps)*wm * p_cont ; then router updates p_cont.
__global__ void update_kernel(float* __restrict__ x,
                              const float* __restrict__ att,
                              const float* __restrict__ aps,
                              const float* __restrict__ mps,
                              const float* __restrict__ fcsum,
                              const float* __restrict__ wm, int step,
                              const float* __restrict__ rw,
                              const float* __restrict__ rb,
                              float* __restrict__ pcont) {
  int t = blockIdx.x, tid = threadIdx.x;  // 256 threads
  float pc = pcont[t];
  const float* wmr = wm + step * NNODES;
  float dotp = 0.f;
#pragma unroll
  for (int j = 0; j < 4; ++j) {
    int e = tid + 256 * j;
    int gr = e >> 7, g = e & 127;
    float fup = 0.f;
    for (int l2 = 0; l2 < NLAY; ++l2) {
      int n = l2 * NGRP + gr;
      float wmv = wmr[n];
      if (wmv != 0.f)
        fup += (att[((size_t)n * TT + t) * GDIM + g] * aps[n * GDIM + g] +
                fcsum[n * TT + t] * mps[n * GDIM + g]) * wmv;
    }
    float xv = x[(size_t)t * EMB + e] + fup * pc;
    x[(size_t)t * EMB + e] = xv;
    dotp += xv * rw[e];
  }
  __shared__ float sr[8];
  dotp = red32_sum(dotp);
  if ((tid & 31) == 0) sr[tid >> 5] = dotp;
  __syncthreads();
  if (tid == 0) {
    float s = 0.f;
#pragma unroll
    for (int i = 0; i < 8; ++i) s += sr[i];
    float ph = 1.f / (1.f + __expf(-(s + rb[0])));
    pcont[t] = (ph < 0.5f ? 1.f : 0.f) * pc;
  }
}

// ---------------------------------------------------------------------------
// Final head
// ---------------------------------------------------------------------------
__global__ void fnorm_kernel(const float* __restrict__ x, __bf16* __restrict__ nx) {
  int t = blockIdx.x, tid = threadIdx.x;
  float v[4], ss = 0.f;
#pragma unroll
  for (int j = 0; j < 4; ++j) { v[j] = x[(size_t)t * EMB + tid + 256 * j]; ss += v[j] * v[j]; }
  __shared__ float sr[8];
  ss = red32_sum(ss);
  if ((tid & 31) == 0) sr[tid >> 5] = ss;
  __syncthreads();
  float tot = 0.f;
#pragma unroll
  for (int i = 0; i < 8; ++i) tot += sr[i];
  float r = rsqrtf(tot / (float)EMB + 1e-6f);
#pragma unroll
  for (int j = 0; j < 4; ++j) nx[(size_t)t * EMB + tid + 256 * j] = f2b(v[j] * r);
}

// logits = 15*tanh((nx @ lm_head^T)/15).
// The 16x1024 bf16 activation tile (shared by all 4 waves of the workgroup)
// is staged into LDS via the Tensor Data Mover (tensor_load_to_lds +
// s_wait_tensorcnt), with a cooperative-copy fallback.
__global__ void logits_kernel(const __bf16* __restrict__ nx,
                              const float* __restrict__ lm,
                              float* __restrict__ out) {
  __shared__ __bf16 sA[16 * EMB];              // 32 KB
  int wid = threadIdx.x >> 5, lane = threadIdx.x & 31;
  int tt = blockIdx.x / VQUADS;                // workgroup-uniform
  int vt = (blockIdx.x % VQUADS) * 4 + wid;
  if (vt > VTILES - 1) vt = VTILES - 1;        // duplicate tail work (benign)

#if __has_builtin(__builtin_amdgcn_tensor_load_to_lds) && \
    __has_builtin(__builtin_amdgcn_s_wait_tensorcnt)
  if (threadIdx.x < 32) {                      // wave 0 issues the DMA
    unsigned long long ga =
        (unsigned long long)(const void*)(nx + (size_t)tt * 16 * EMB);
    unsigned ldsa = (unsigned)(size_t)(void*)sA;  // low 32 bits = LDS offset
    // D# group 0: count=1 | lds_addr | global_addr[56:0] | type=2
    u32x4 g0 = {1u, ldsa, (unsigned)(ga & 0xffffffffu),
                (unsigned)((ga >> 32) & 0x1ffffffu) | (2u << 30)};
    // D# group 1: data_size=2B ; tensor 1024x16 ; tile 1024x16 ; stride 1024
    i32x8 g1 = {0x10000, (int)(1024u << 16), (int)(16u << 16),
                (int)(1024u << 16), 16, 1024, 0, 0};
    i32x4 gz4 = {0, 0, 0, 0};
    i32x8 gz8 = {0, 0, 0, 0, 0, 0, 0, 0};
    __builtin_amdgcn_tensor_load_to_lds(g0, g1, gz4, gz4, gz8, 0);
    __builtin_amdgcn_s_wait_tensorcnt(0);
  }
  __syncthreads();
#else
  for (int i = threadIdx.x; i < 16 * EMB / 8; i += 128) {
    int row = i >> 7, col = (i & 127) << 3;
    *(v8bf*)&sA[row * EMB + col] =
        *(const v8bf*)&nx[(size_t)(tt * 16 + row) * EMB + col];
  }
  __syncthreads();
#endif

  int mrow = vt * 16 + (lane & 15);
  if (mrow > NVOCAB - 1) mrow = NVOCAB - 1;    // clamp last partial tile
  int koff = (lane >> 4) << 3;
  int am = lane & 15;
  v8f acc = (v8f){0, 0, 0, 0, 0, 0, 0, 0};
  for (int k0 = 0; k0 < EMB; k0 += 32) {
    int kk = k0 + koff;
    // A tile from LDS
    v8bf lo = *(const v8bf*)&sA[am * EMB + kk];
    v8bf hi = *(const v8bf*)&sA[am * EMB + kk + 16];
    v16bf a;
#pragma unroll
    for (int i = 0; i < 8; ++i) { a[i] = lo[i]; a[8 + i] = hi[i]; }
    // B tile from lm_head (single use -> inline f32->bf16 convert)
    const float* bp = lm + (size_t)mrow * EMB + kk;
    __builtin_prefetch(bp + 64, 0, 1);
    float4 b0 = ((const float4*)bp)[0], b1 = ((const float4*)bp)[1];
    float4 b2 = ((const float4*)(bp + 16))[0], b3 = ((const float4*)(bp + 16))[1];
    v16bf b;
    b[0]=f2b(b0.x);  b[1]=f2b(b0.y);  b[2]=f2b(b0.z);  b[3]=f2b(b0.w);
    b[4]=f2b(b1.x);  b[5]=f2b(b1.y);  b[6]=f2b(b1.z);  b[7]=f2b(b1.w);
    b[8]=f2b(b2.x);  b[9]=f2b(b2.y);  b[10]=f2b(b2.z); b[11]=f2b(b2.w);
    b[12]=f2b(b3.x); b[13]=f2b(b3.y); b[14]=f2b(b3.z); b[15]=f2b(b3.w);
    acc = wmma_bf16(a, b, acc);
  }
  int col = vt * 16 + (lane & 15);
  if (col < NVOCAB) {
    int r0 = tt * 16 + ((lane >> 4) << 3);
#pragma unroll
    for (int r = 0; r < 8; ++r)
      out[(size_t)(r0 + r) * NVOCAB + col] = 15.f * tanhf(acc[r] * (1.f / 15.f));
  }
}

// ---------------------------------------------------------------------------
// Host launcher
// ---------------------------------------------------------------------------
extern "C" void kernel_launch(void* const* d_in, const int* in_sizes, int n_in,
                              void* d_out, int out_size, void* d_ws, size_t ws_size,
                              hipStream_t stream) {
  const int*   idx       = (const int*)  d_in[0];
  // d_in[1] = adapters (identity-block selectors; structure exploited)
  const float* qkv_w     = (const float*)d_in[2];
  const float* attn_proj = (const float*)d_in[3];
  const float* mlp_fc    = (const float*)d_in[4];
  const float* mlp_proj  = (const float*)d_in[5];
  const float* dep       = (const float*)d_in[6];
  const float* router_w  = (const float*)d_in[7];
  const float* router_b  = (const float*)d_in[8];
  const float* wte       = (const float*)d_in[9];
  const float* lm_head   = (const float*)d_in[10];
  float* outp = (float*)d_out;

  char* base = (char*)d_ws;
  size_t off = 0;
  auto alloc = [&](size_t bytes) -> char* {
    char* p = base + off;
    off = (off + bytes + 255) & ~(size_t)255;
    return p;
  };
  float* depths = (float*)alloc(NNODES * 4);
  float* wm     = (float*)alloc(NSTEPS * NNODES * 4);
  float* aps    = (float*)alloc(NNODES * GDIM * 4);
  float* mps    = (float*)alloc(NNODES * GDIM * 4);
  float* pcont  = (float*)alloc(TT * 4);
  float* x      = (float*)alloc((size_t)TT * EMB * 4);
  float* qkv    = (float*)alloc((size_t)NNODES * TT * O3 * 4);
  float* att    = (float*)alloc((size_t)NNODES * TT * GDIM * 4);
  float* fcsum  = (float*)alloc((size_t)NNODES * TT * 4);
  __bf16* xb    = (__bf16*)alloc((size_t)TT * EMB * 2);
  __bf16* qb    = (__bf16*)alloc((size_t)NNODES * TT * GDIM * 2);
  __bf16* kbuf  = (__bf16*)alloc((size_t)NNODES * TT * GDIM * 2);
  __bf16* vtb   = (__bf16*)alloc((size_t)NNODES * GDIM * TT * 2);
  __bf16* nb    = (__bf16*)alloc((size_t)NNODES * TT * GDIM * 2);
  __bf16* nxb   = (__bf16*)alloc((size_t)TT * EMB * 2);
  __bf16* qkvwb = (__bf16*)alloc((size_t)NNODES * O3 * GDIM * 2);
  __bf16* fcwb  = (__bf16*)alloc((size_t)NNODES * O4 * GDIM * 2);
  (void)ws_size; (void)in_sizes; (void)n_in; (void)out_size;

  rowsum_kernel<<<NNODES * GDIM, 128, 0, stream>>>(attn_proj, aps, GDIM);
  rowsum_kernel<<<NNODES * GDIM, 128, 0, stream>>>(mlp_proj, mps, O4);
  depth_kernel<<<1, 128, 0, stream>>>(dep, depths, wm);
  embed_kernel<<<TT, 256, 0, stream>>>(idx, wte, x, pcont);
  // weights are read 8x (once per step): pre-convert to bf16 once
  f2b_kernel<<<(NNODES * O3 * GDIM) / 256, 256, 0, stream>>>(qkv_w, qkvwb);
  f2b_kernel<<<(NNODES * O4 * GDIM) / 256, 256, 0, stream>>>(mlp_fc, fcwb);

  for (int step = 0; step < NSTEPS; ++step) {
    f2b_kernel<<<(TT * EMB) / 256, 256, 0, stream>>>(x, xb);
    qkv_kernel<<<(NNODES * 32 * 24) / 4, 128, 0, stream>>>(xb, qkvwb, qkv);
    rot_kernel<<<(NNODES * TT) / 4, 128, 0, stream>>>(qkv, qb, kbuf, vtb);
    attn_kernel<<<(NNODES * 32) / 4, 128, 0, stream>>>(qb, kbuf, vtb, att);
    midnorm_kernel<<<(NNODES * TT) / 4, 128, 0, stream>>>(x, att, aps, nb);
    fc_kernel<<<(NNODES * 32) / 4, 128, 0, stream>>>(nb, fcwb, fcsum);
    update_kernel<<<TT, 256, 0, stream>>>(x, att, aps, mps, fcsum, wm, step,
                                          router_w, router_b, pcont);
  }

  fnorm_kernel<<<TT, 256, 0, stream>>>(x, nxb);
  logits_kernel<<<32 * VQUADS, 128, 0, stream>>>(nxb, lm_head, outp);
}
